// AggrHGraphConvWindow_79285096284407
// MI455X (gfx1250) — matrix operand order repaired and checked by reading, compiled, not run on hardware
//
#include <hip/hip_runtime.h>
#include <hip/hip_bf16.h>
#include <stdint.h>

#define N_NODE 100
#define N_POD  1500
#define N_SVC  400
#define NTOT   2000
#define TT     32
#define FF     64
#define IND    128
#define HHH    256
#define G4H    1024
#define E_SVC  1600
#define E_IN   1500
#define E_NI   1500

typedef __attribute__((ext_vector_type(16))) _Float16 v16h;
typedef __attribute__((ext_vector_type(8)))  _Float16 v8h;
typedef __attribute__((ext_vector_type(8)))  float    v8f;
typedef __attribute__((ext_vector_type(4)))  float    v4f;

union V16U { v16h v; v8h h[2]; _Float16 e[16]; };

// A-fragment (16x32 f16, M x K): lane L holds row L&15; contiguous K run of 16
// starting at k-offset (lane<16?0:16). Two 16B loads.
__device__ __forceinline__ v16h load_a_frag(const _Float16* rowbase_k, int lane) {
  const _Float16* q = rowbase_k + ((lane & 16) ? 16 : 0);
  V16U u;
  u.h[0] = *(const v8h*)(q);
  u.h[1] = *(const v8h*)(q + 8);
  return u.v;
}

// B-fragment (32x16 f16, K x N) from a [N,K] row-major weight: lane holds
// column n = lane&15, contiguous K run starting at k0 + (lane<16?0:16).
__device__ __forceinline__ v16h load_b_frag_NK(const _Float16* wrow_k, int lane) {
  return *(const v16h*)(wrow_k + ((lane & 16) ? 16 : 0));
}

__device__ __forceinline__ float sigmoidf_(float x) { return 1.0f / (1.0f + expf(-x)); }

// ---------------------------------------------------------------- utilities
__global__ void zero_u32_kernel(uint32_t* __restrict__ p, long n) {
  long i = (long)blockIdx.x * blockDim.x + threadIdx.x;
  if (i < n) p[i] = 0u;
}

__global__ void f32_to_f16_kernel(const float* __restrict__ s, _Float16* __restrict__ d, long n) {
  long i = (long)blockIdx.x * blockDim.x + threadIdx.x;
  if (i < n) d[i] = (_Float16)s[i];
}

// Convert + transpose conv weights: W [T, F, IND] f32 -> Wt [T, IND, F] f16
// so that conv B-fragments become lane-contiguous ([N,K] layout).
__global__ void convw_transpose_kernel(const float* __restrict__ s, _Float16* __restrict__ d) {
  long i = (long)blockIdx.x * blockDim.x + threadIdx.x;   // over T*F*IND
  if (i >= (long)TT * FF * IND) return;
  int t = (int)(i >> 13);          // FF*IND = 8192
  int r = (int)(i & 8191);
  int k = r >> 7;                  // f index
  int n = r & 127;                 // out-channel
  d[(size_t)t * IND * FF + (size_t)n * FF + k] = (_Float16)s[i];
}

// ---------------------------------------------------------------- graph conv
__global__ void degree_kernel(const int* __restrict__ src, const int* __restrict__ dst,
                              int ne, float* __restrict__ dout, float* __restrict__ din) {
  int e = blockIdx.x * blockDim.x + threadIdx.x;
  if (e < ne) {
    atomicAdd(&dout[src[e]], 1.0f);
    atomicAdd(&din[dst[e]], 1.0f);
  }
}

// agg[dst] += feat[src] * rsqrt(max(deg_out[src],1)) * rsqrt(max(deg_in[dst],1))
__global__ void aggregate_kernel(const float* __restrict__ feat,
                                 const int* __restrict__ src, const int* __restrict__ dst,
                                 int ne, const float* __restrict__ deg_out,
                                 const float* __restrict__ deg_in,
                                 float* __restrict__ aggdst) {
  int tid = blockIdx.x * blockDim.x + threadIdx.x;
  int e = tid >> 11;            // TT*FF = 2048 elements per edge
  int j = tid & 2047;
  if (e >= ne) return;
  int s = src[e], d = dst[e];
  float w = rsqrtf(fmaxf(deg_out[s], 1.0f)) * rsqrtf(fmaxf(deg_in[d], 1.0f));
  atomicAdd(&aggdst[(size_t)d * 2048 + j], feat[(size_t)s * 2048 + j] * w);
}

// x[row0+m, t, :] = leaky_relu( agg[row0+m, t, :64] @ W[t] + b[t] ), output f16.
// Block = 256 threads (8 waves); wave w owns n-tile [16w,16w+16); grid.z = t.
// Wt is the transposed f16 weight [T, IND, F] ([N,K] layout).
__global__ __launch_bounds__(256)
void conv_gemm_kernel(const float* __restrict__ agg, int row0, int nrows,
                      const _Float16* __restrict__ Wt,  // [T, IND, FF] f16
                      const float* __restrict__ bias,   // [T, IND]
                      _Float16* __restrict__ x) {       // [2000, 32, IND] f16
  const int t     = blockIdx.z;
  const int mbase = blockIdx.x * 16;
  const int wave  = threadIdx.x >> 5;
  const int lane  = threadIdx.x & 31;
  const int nbase = wave * 16;
  const int col   = lane & 15;
  const int arow  = lane & 15;
  const float* abase = agg + ((size_t)(row0 + mbase + arow) * TT + t) * FF;
  const _Float16* wrow = Wt + (size_t)t * IND * FF + (size_t)(nbase + col) * FF;
  v8f acc = {};
  #pragma unroll
  for (int k0 = 0; k0 < FF; k0 += 32) {
    const float* ap = abase + k0 + ((lane & 16) ? 16 : 0);
    v4f a0 = *(const v4f*)(ap);
    v4f a1 = *(const v4f*)(ap + 4);
    v4f a2 = *(const v4f*)(ap + 8);
    v4f a3 = *(const v4f*)(ap + 12);
    V16U ua;
    #pragma unroll
    for (int i = 0; i < 4; ++i) {
      ua.e[i]      = (_Float16)a0[i];
      ua.e[4 + i]  = (_Float16)a1[i];
      ua.e[8 + i]  = (_Float16)a2[i];
      ua.e[12 + i] = (_Float16)a3[i];
    }
    v16h bf = load_b_frag_NK(wrow + k0, lane);
    acc = __builtin_amdgcn_wmma_f32_16x16x32_f16(false, ua.v, false, bf,
                                                 (short)0, acc, false, false);
  }
  const float bv = bias[t * IND + nbase + col];
  #pragma unroll
  for (int r = 0; r < 8; ++r) {
    const int mi = mbase + r + ((lane & 16) ? 8 : 0);
    if (mi < nrows) {
      float v = acc[r] + bv;
      v = (v > 0.0f) ? v : 0.01f * v;
      x[((size_t)(row0 + mi) * TT + t) * IND + nbase + col] = (_Float16)v;
    }
  }
}

// ---------------------------------------------------------------- LSTM step
// g[m, n] = sum_k A1[m,k] Wih[n,k] + sum_k A2[m,k] Whh[n,k]
// A1: layer input at time t (f16, row stride lda1), K1 = 128 or 256
// A2: h_prev (f16, row stride lda2), K2 = 256.  M = 2000 = 16*125 exactly.
// Each wave computes a 16x64 strip (4 n-tiles) to reuse A-fragments 4x.
// Grid (125, 2), block 256 = 8 waves: strip index = blockIdx.y*8 + wave.
__global__ __launch_bounds__(256)
void lstm_gemm_kernel(const _Float16* __restrict__ A1, int lda1, int K1,
                      const _Float16* __restrict__ A2, int lda2,
                      const _Float16* __restrict__ Wih,   // [1024, K1] row-major
                      const _Float16* __restrict__ Whh,   // [1024, 256] row-major
                      float* __restrict__ g) {            // [2000, 1024]
  const int mbase = blockIdx.x * 16;
  const int strip = blockIdx.y * 8 + (threadIdx.x >> 5);  // 0..15
  const int lane  = threadIdx.x & 31;
  const int nbase = strip * 64;
  const int arow  = lane & 15;
  const int col   = lane & 15;
  v8f acc0 = {}, acc1 = {}, acc2 = {}, acc3 = {};
  {
    const _Float16* a  = A1 + (size_t)(mbase + arow) * lda1;
    const _Float16* w0 = Wih + (size_t)(nbase + col) * K1;
    const _Float16* w1 = w0 + (size_t)16 * K1;
    const _Float16* w2 = w0 + (size_t)32 * K1;
    const _Float16* w3 = w0 + (size_t)48 * K1;
    for (int k0 = 0; k0 < K1; k0 += 32) {
      v16h af = load_a_frag(a + k0, lane);
      acc0 = __builtin_amdgcn_wmma_f32_16x16x32_f16(false, af, false,
               load_b_frag_NK(w0 + k0, lane), (short)0, acc0, false, false);
      acc1 = __builtin_amdgcn_wmma_f32_16x16x32_f16(false, af, false,
               load_b_frag_NK(w1 + k0, lane), (short)0, acc1, false, false);
      acc2 = __builtin_amdgcn_wmma_f32_16x16x32_f16(false, af, false,
               load_b_frag_NK(w2 + k0, lane), (short)0, acc2, false, false);
      acc3 = __builtin_amdgcn_wmma_f32_16x16x32_f16(false, af, false,
               load_b_frag_NK(w3 + k0, lane), (short)0, acc3, false, false);
    }
  }
  {
    const _Float16* a  = A2 + (size_t)(mbase + arow) * lda2;
    const _Float16* w0 = Whh + (size_t)(nbase + col) * HHH;
    const _Float16* w1 = w0 + (size_t)16 * HHH;
    const _Float16* w2 = w0 + (size_t)32 * HHH;
    const _Float16* w3 = w0 + (size_t)48 * HHH;
    #pragma unroll
    for (int k0 = 0; k0 < HHH; k0 += 32) {
      v16h af = load_a_frag(a + k0, lane);
      acc0 = __builtin_amdgcn_wmma_f32_16x16x32_f16(false, af, false,
               load_b_frag_NK(w0 + k0, lane), (short)0, acc0, false, false);
      acc1 = __builtin_amdgcn_wmma_f32_16x16x32_f16(false, af, false,
               load_b_frag_NK(w1 + k0, lane), (short)0, acc1, false, false);
      acc2 = __builtin_amdgcn_wmma_f32_16x16x32_f16(false, af, false,
               load_b_frag_NK(w2 + k0, lane), (short)0, acc2, false, false);
      acc3 = __builtin_amdgcn_wmma_f32_16x16x32_f16(false, af, false,
               load_b_frag_NK(w3 + k0, lane), (short)0, acc3, false, false);
    }
  }
  #pragma unroll
  for (int r = 0; r < 8; ++r) {
    const int mi = mbase + r + ((lane & 16) ? 8 : 0);
    float* gp = g + (size_t)mi * G4H + nbase + col;
    gp[0]  = acc0[r];
    gp[16] = acc1[r];
    gp[32] = acc2[r];
    gp[48] = acc3[r];
  }
}

// Gate nonlinearity + state update. PyTorch gate order i, f, g, o.
__global__ void lstm_pointwise_kernel(const float* __restrict__ g,
                                      const float* __restrict__ b_ih,
                                      const float* __restrict__ b_hh,
                                      float* __restrict__ c,
                                      _Float16* __restrict__ hprev,
                                      _Float16* __restrict__ h16_out, size_t h16_stride,
                                      float* __restrict__ f32_out,    size_t f32_stride) {
  int idx = blockIdx.x * blockDim.x + threadIdx.x;
  if (idx >= NTOT * HHH) return;
  int m = idx >> 8;
  int n = idx & 255;
  const float* gr = g + (size_t)m * G4H;
  float i = sigmoidf_(gr[n]           + b_ih[n]           + b_hh[n]);
  float f = sigmoidf_(gr[HHH + n]     + b_ih[HHH + n]     + b_hh[HHH + n]);
  float gg = tanhf(   gr[2 * HHH + n] + b_ih[2 * HHH + n] + b_hh[2 * HHH + n]);
  float o = sigmoidf_(gr[3 * HHH + n] + b_ih[3 * HHH + n] + b_hh[3 * HHH + n]);
  float cc = f * c[idx] + i * gg;
  c[idx] = cc;
  float h = o * tanhf(cc);
  hprev[idx] = (_Float16)h;
  if (h16_out) h16_out[(size_t)m * h16_stride + n] = (_Float16)h;
  if (f32_out) f32_out[(size_t)m * f32_stride + n] = h;
}

// ---------------------------------------------------------------- launch
static inline size_t align_up_(size_t x, size_t a) { return (x + a - 1) & ~(a - 1); }

extern "C" void kernel_launch(void* const* d_in, const int* in_sizes, int n_in,
                              void* d_out, int out_size, void* d_ws, size_t ws_size,
                              hipStream_t stream) {
  (void)in_sizes; (void)n_in; (void)out_size; (void)ws_size;

  const float* node_feat = (const float*)d_in[0];
  const float* pod_feat  = (const float*)d_in[1];
  const float* svc_feat  = (const float*)d_in[2];
  const float* W_svc = (const float*)d_in[3];
  const float* b_svc = (const float*)d_in[4];
  const float* W_in  = (const float*)d_in[5];
  const float* b_in  = (const float*)d_in[6];
  const float* W_ni  = (const float*)d_in[7];
  const float* b_ni  = (const float*)d_in[8];
  const float* W_ih0 = (const float*)d_in[9];
  const float* W_hh0 = (const float*)d_in[10];
  const float* b_ih0 = (const float*)d_in[11];
  const float* b_hh0 = (const float*)d_in[12];
  const float* W_ih1 = (const float*)d_in[13];
  const float* W_hh1 = (const float*)d_in[14];
  const float* b_ih1 = (const float*)d_in[15];
  const float* b_hh1 = (const float*)d_in[16];
  const int* svc_src = (const int*)d_in[17];
  const int* svc_dst = (const int*)d_in[18];
  const int* in_src  = (const int*)d_in[19];
  const int* in_dst  = (const int*)d_in[20];
  const int* ni_src  = (const int*)d_in[21];
  const int* ni_dst  = (const int*)d_in[22];
  float* out = (float*)d_out;

  // Bump-allocate workspace (256B aligned regions).
  char* ws = (char*)d_ws;
  size_t off = 0;
  auto alloc = [&](size_t bytes) -> void* {
    off = align_up_(off, 256);
    void* p = ws + off;
    off += bytes;
    return p;
  };

  float*    agg   = (float*)   alloc((size_t)NTOT * TT * FF * 4);   // 16.38 MB
  float*    deg   = (float*)   alloc(4000 * 4);                     // contiguous after agg
  float*    gbuf  = (float*)   alloc((size_t)NTOT * G4H * 4);       // 8.19 MB
  float*    cbuf  = (float*)   alloc((size_t)NTOT * HHH * 4);       // 2.05 MB
  _Float16* xh    = (_Float16*)alloc((size_t)NTOT * TT * IND * 2);  // 16.38 MB
  _Float16* h1h   = (_Float16*)alloc((size_t)NTOT * TT * HHH * 2);  // 32.77 MB
  _Float16* hprev = (_Float16*)alloc((size_t)NTOT * HHH * 2);       // 1.02 MB
  _Float16* Wsvc_h = (_Float16*)alloc((size_t)TT * IND * FF * 2);   // transposed [T,IND,F]
  _Float16* Win_h  = (_Float16*)alloc((size_t)TT * IND * FF * 2);
  _Float16* Wni_h  = (_Float16*)alloc((size_t)TT * IND * FF * 2);
  _Float16* Wih0_h = (_Float16*)alloc((size_t)G4H * IND * 2);
  _Float16* Whh0_h = (_Float16*)alloc((size_t)G4H * HHH * 2);
  _Float16* Wih1_h = (_Float16*)alloc((size_t)G4H * HHH * 2);
  _Float16* Whh1_h = (_Float16*)alloc((size_t)G4H * HHH * 2);

  float* deg_svc_o = deg;         float* deg_svc_i = deg + 400;
  float* deg_in_o  = deg + 800;   float* deg_in_i  = deg + 2300;  // 1500 pods / 100 nodes
  float* deg_ni_o  = deg + 2400;  float* deg_ni_i  = deg + 2500;  // 100 nodes / 1500 pods

  const int BS = 256;
  auto blocks = [](long n, int bs) { return (unsigned)((n + bs - 1) / bs); };

  // 1) zero agg + degrees (contiguous span)
  {
    long nzero = (long)NTOT * TT * FF + 4000;
    zero_u32_kernel<<<blocks(nzero, BS), BS, 0, stream>>>((uint32_t*)agg, nzero);
  }

  // 2) weights -> f16 (conv weights also transposed to [N,K] layout)
  convw_transpose_kernel<<<blocks((long)TT*FF*IND, BS), BS, 0, stream>>>(W_svc, Wsvc_h);
  convw_transpose_kernel<<<blocks((long)TT*FF*IND, BS), BS, 0, stream>>>(W_in,  Win_h);
  convw_transpose_kernel<<<blocks((long)TT*FF*IND, BS), BS, 0, stream>>>(W_ni,  Wni_h);
  f32_to_f16_kernel<<<blocks((long)G4H*IND, BS), BS, 0, stream>>>(W_ih0, Wih0_h, (long)G4H*IND);
  f32_to_f16_kernel<<<blocks((long)G4H*HHH, BS), BS, 0, stream>>>(W_hh0, Whh0_h, (long)G4H*HHH);
  f32_to_f16_kernel<<<blocks((long)G4H*HHH, BS), BS, 0, stream>>>(W_ih1, Wih1_h, (long)G4H*HHH);
  f32_to_f16_kernel<<<blocks((long)G4H*HHH, BS), BS, 0, stream>>>(W_hh1, Whh1_h, (long)G4H*HHH);

  // 3) degrees
  degree_kernel<<<blocks(E_SVC, BS), BS, 0, stream>>>(svc_src, svc_dst, E_SVC, deg_svc_o, deg_svc_i);
  degree_kernel<<<blocks(E_IN,  BS), BS, 0, stream>>>(in_src,  in_dst,  E_IN,  deg_in_o,  deg_in_i);
  degree_kernel<<<blocks(E_NI,  BS), BS, 0, stream>>>(ni_src,  ni_dst,  E_NI,  deg_ni_o,  deg_ni_i);

  // 4) normalized aggregation; x row order = [node(100), pod(1500), svc(400)]
  aggregate_kernel<<<blocks((long)E_IN  * 2048, BS), BS, 0, stream>>>(
      pod_feat, in_src, in_dst, E_IN, deg_in_o, deg_in_i, agg);                        // node rows
  aggregate_kernel<<<blocks((long)E_NI  * 2048, BS), BS, 0, stream>>>(
      node_feat, ni_src, ni_dst, E_NI, deg_ni_o, deg_ni_i, agg + (size_t)100 * 2048);  // pod rows
  aggregate_kernel<<<blocks((long)E_SVC * 2048, BS), BS, 0, stream>>>(
      svc_feat, svc_src, svc_dst, E_SVC, deg_svc_o, deg_svc_i, agg + (size_t)1600 * 2048); // svc

  // 5) per-timestep conv GEMM (+bias, leaky ReLU) -> f16 x
  conv_gemm_kernel<<<dim3(7, 1, TT),  BS, 0, stream>>>(agg, 0,    N_NODE, Win_h,  b_in,  xh);
  conv_gemm_kernel<<<dim3(94, 1, TT), BS, 0, stream>>>(agg, 100,  N_POD,  Wni_h,  b_ni,  xh);
  conv_gemm_kernel<<<dim3(25, 1, TT), BS, 0, stream>>>(agg, 1600, N_SVC,  Wsvc_h, b_svc, xh);

  // 6) LSTM layer 0 : x[2000,32,128] -> h1h[2000,32,256] (f16)
  zero_u32_kernel<<<blocks((long)NTOT * HHH,     BS), BS, 0, stream>>>((uint32_t*)cbuf,  (long)NTOT * HHH);
  zero_u32_kernel<<<blocks((long)NTOT * HHH / 2, BS), BS, 0, stream>>>((uint32_t*)hprev, (long)NTOT * HHH / 2);
  for (int t = 0; t < TT; ++t) {
    lstm_gemm_kernel<<<dim3(125, 2), BS, 0, stream>>>(
        xh + (size_t)t * IND, TT * IND, IND, hprev, HHH, Wih0_h, Whh0_h, gbuf);
    lstm_pointwise_kernel<<<blocks((long)NTOT * HHH, BS), BS, 0, stream>>>(
        gbuf, b_ih0, b_hh0, cbuf, hprev,
        h1h + (size_t)t * HHH, (size_t)TT * HHH, (float*)nullptr, 0);
  }

  // 7) LSTM layer 1 : h1h -> d_out[2000,32,256] (f32)
  zero_u32_kernel<<<blocks((long)NTOT * HHH,     BS), BS, 0, stream>>>((uint32_t*)cbuf,  (long)NTOT * HHH);
  zero_u32_kernel<<<blocks((long)NTOT * HHH / 2, BS), BS, 0, stream>>>((uint32_t*)hprev, (long)NTOT * HHH / 2);
  for (int t = 0; t < TT; ++t) {
    lstm_gemm_kernel<<<dim3(125, 2), BS, 0, stream>>>(
        h1h + (size_t)t * HHH, TT * HHH, HHH, hprev, HHH, Wih1_h, Whh1_h, gbuf);
    lstm_pointwise_kernel<<<blocks((long)NTOT * HHH, BS), BS, 0, stream>>>(
        gbuf, b_ih1, b_hh1, cbuf, hprev,
        (_Float16*)nullptr, 0, out + (size_t)t * HHH, (size_t)TT * HHH);
  }
}
